// Decoder_59365037965431
// MI455X (gfx1250) — matrix operand order repaired and checked by reading, not compiled, not run on hardware
//
#include <hip/hip_runtime.h>

#define B_ 32
#define S_ 64
#define T_ 64
#define EMB_ 256
#define HID_ 512
#define VOCAB_ 32000
#define H3_ (3 * HID_)
#define XD_ (EMB_ + HID_)   // 768: concat(emb, ctx)
#define DC_ (2 * HID_)      // 1024: concat(dec, ctx)

typedef __attribute__((ext_vector_type(16))) __bf16 v16bf;
typedef __attribute__((ext_vector_type(8)))  __bf16 v8bf;
typedef __attribute__((ext_vector_type(8)))  float  v8f;

__device__ __forceinline__ unsigned short f2bf(float x) {
  unsigned int u = __float_as_uint(x);
  u += 0x7FFFu + ((u >> 16) & 1u);   // round-to-nearest-even
  return (unsigned short)(u >> 16);
}

__global__ void zero_u32_kernel(unsigned int* __restrict__ p, long long n) {
  long long i = (long long)blockIdx.x * blockDim.x + threadIdx.x;
  if (i < n) p[i] = 0u;
}

__global__ void f32_to_bf16_kernel(const float* __restrict__ in,
                                   unsigned short* __restrict__ out, long long n) {
  long long i  = (long long)blockIdx.x * blockDim.x + threadIdx.x;
  long long st = (long long)gridDim.x * blockDim.x;
  for (; i < n; i += st) out[i] = f2bf(in[i]);
}

// C[M x N] = A[M x K] (bf16, row stride lda) @ W[N x K]^T (bf16) + bias
// fp32 accumulation via V_WMMA_F32_16X16X32_BF16.
// One wave per 32-wide N tile (two B fragments reuse the same two A fragments:
// 4 WMMAs per 8 b128 loads), 32 rows of M per blockIdx.y.
__global__ void gemm_bf16_wmma(const unsigned short* __restrict__ A, int lda,
                               const unsigned short* __restrict__ W,
                               const float* __restrict__ bias,
                               float* __restrict__ C, long long ldc,
                               int N, int K) {
  const int lane = threadIdx.x & 31;
  const int wave = threadIdx.x >> 5;
  const int tile = blockIdx.x * (blockDim.x >> 5) + wave;
  const int n0   = tile * 32;
  if (n0 >= N) return;                  // wave-uniform: EXEC stays all-ones for WMMA
  const int m0 = blockIdx.y * 32;
  const int lh = lane & 15;             // column / row-in-tile selector
  const int hi = lane >> 4;             // half-wave selector

  // B fragment (K x N): lane holds col n+lh, contiguous K range [hi*16, hi*16+16)
  const unsigned short* Wp0 = W + (size_t)(n0 + lh) * K + hi * 16;
  const unsigned short* Wp1 = W + (size_t)(n0 + 16 + lh) * K + hi * 16;
  // A fragment (M x K): lane holds row m, contiguous K range [hi*16, hi*16+16)
  const unsigned short* Ap0 = A + (size_t)(m0 + lh) * lda + hi * 16;
  const unsigned short* Ap1 = A + (size_t)(m0 + 16 + lh) * lda + hi * 16;

  v8f acc00 = {};   // rows m0..m0+15, cols n0..n0+15
  v8f acc01 = {};   // rows m0..m0+15, cols n0+16..n0+31
  v8f acc10 = {};   // rows m0+16..m0+31, cols n0..n0+15
  v8f acc11 = {};   // rows m0+16..m0+31, cols n0+16..n0+31
  for (int k0 = 0; k0 < K; k0 += 32) {
    v16bf b0 = *reinterpret_cast<const v16bf*>(Wp0 + k0);
    v16bf b1 = *reinterpret_cast<const v16bf*>(Wp1 + k0);
    v8bf a0lo = *reinterpret_cast<const v8bf*>(Ap0 + k0);
    v8bf a0hi = *reinterpret_cast<const v8bf*>(Ap0 + k0 + 8);
    v8bf a1lo = *reinterpret_cast<const v8bf*>(Ap1 + k0);
    v8bf a1hi = *reinterpret_cast<const v8bf*>(Ap1 + k0 + 8);
    v16bf a0, a1;
#pragma unroll
    for (int i = 0; i < 8; ++i) {
      a0[i] = a0lo[i]; a0[i + 8] = a0hi[i];
      a1[i] = a1lo[i]; a1[i + 8] = a1hi[i];
    }
    acc00 = __builtin_amdgcn_wmma_f32_16x16x32_bf16(false, a0, false, b0,
                                                    (short)0, acc00, false, false);
    acc01 = __builtin_amdgcn_wmma_f32_16x16x32_bf16(false, a0, false, b1,
                                                    (short)0, acc01, false, false);
    acc10 = __builtin_amdgcn_wmma_f32_16x16x32_bf16(false, a1, false, b0,
                                                    (short)0, acc10, false, false);
    acc11 = __builtin_amdgcn_wmma_f32_16x16x32_bf16(false, a1, false, b1,
                                                    (short)0, acc11, false, false);
  }
  const float bv0 = bias ? bias[n0 + lh] : 0.0f;
  const float bv1 = bias ? bias[n0 + 16 + lh] : 0.0f;
#pragma unroll
  for (int r = 0; r < 8; ++r) {
    int m = m0 + r + (hi << 3);   // D layout: VGPR r -> row r (lanes 0-15) / 8+r (16-31)
    C[(size_t)m        * ldc + n0 + lh]      = acc00[r] + bv0;
    C[(size_t)m        * ldc + n0 + 16 + lh] = acc01[r] + bv1;
    C[(size_t)(m + 16) * ldc + n0 + lh]      = acc10[r] + bv0;
    C[(size_t)(m + 16) * ldc + n0 + 16 + lh] = acc11[r] + bv1;
  }
}

// x_bf[b, 0:256] = emb_table[tgt_in[b,t]], x_bf[b, 256:768] = ctx[b]  (bf16)
__global__ void embed_concat_kernel(const float* __restrict__ emb_table,
                                    const int* __restrict__ tgt_in,
                                    const float* __restrict__ ctx,
                                    unsigned short* __restrict__ x_bf, int t) {
  int i = blockIdx.x * blockDim.x + threadIdx.x;
  if (i >= B_ * XD_) return;
  int b = i / XD_, c = i - b * XD_;
  float v;
  if (c < EMB_) {
    int tok = tgt_in[b * T_ + t];
    v = emb_table[(size_t)tok * EMB_ + c];
  } else {
    v = ctx[b * HID_ + (c - EMB_)];
  }
  x_bf[i] = f2bf(v);
}

// PyTorch GRUCell gate fusion. h_bf written with row stride ld_bf (bf16 copy
// feeding the next WMMA GEMM as its A matrix).
__global__ void gru_combine_kernel(const float* __restrict__ gi,
                                   const float* __restrict__ gh,
                                   const float* __restrict__ h_prev,
                                   float* __restrict__ h_out,
                                   unsigned short* __restrict__ h_bf, int ld_bf) {
  int i = blockIdx.x * blockDim.x + threadIdx.x;
  if (i >= B_ * HID_) return;
  int b = i >> 9, j = i & (HID_ - 1);
  const float* gib = gi + (size_t)b * H3_;
  const float* ghb = gh + (size_t)b * H3_;
  float r  = 1.0f / (1.0f + __expf(-(gib[j] + ghb[j])));
  float z  = 1.0f / (1.0f + __expf(-(gib[HID_ + j] + ghb[HID_ + j])));
  float nn = tanhf(gib[2 * HID_ + j] + r * ghb[2 * HID_ + j]);
  float h  = (1.0f - z) * nn + z * h_prev[i];
  h_out[i] = h;
  h_bf[(size_t)b * ld_bf + j] = f2bf(h);
}

// Bahdanau attention, one block (8 waves) per batch row.
// e_s = v . tanh(q + kproj[s]); softmax over S (masked); ctx = sum a_s * enc_out[s]
__global__ void attention_kernel(const float* __restrict__ q,
                                 const float* __restrict__ kproj,
                                 const float* __restrict__ enc_out,
                                 const unsigned char* __restrict__ src_mask,
                                 const float* __restrict__ vvec,
                                 float* __restrict__ ctx_f32,
                                 unsigned short* __restrict__ dec_ctx_bf) {
  __shared__ float e_sm[S_];
  __shared__ float a_sm[S_];
  const int b    = blockIdx.x;
  const int wave = threadIdx.x >> 5;
  const int lane = threadIdx.x & 31;
  const float* qb = q + (size_t)b * HID_;

  for (int s = wave; s < S_; s += 8) {
    const float* kp = kproj + ((size_t)b * S_ + s) * HID_;
    float part = 0.0f;
    for (int h = lane; h < HID_; h += 32)
      part += vvec[h] * tanhf(qb[h] + kp[h]);
    for (int off = 16; off > 0; off >>= 1) part += __shfl_down(part, off, 32);
    if (lane == 0)
      e_sm[s] = src_mask[b * S_ + s] ? -1e9f : part;
  }
  __syncthreads();

  if (wave == 0) {
    float x0 = e_sm[lane], x1 = e_sm[lane + 32];
    float m = fmaxf(x0, x1);
    for (int off = 16; off > 0; off >>= 1) m = fmaxf(m, __shfl_xor(m, off, 32));
    float e0 = __expf(x0 - m), e1 = __expf(x1 - m);
    float sum = e0 + e1;
    for (int off = 16; off > 0; off >>= 1) sum += __shfl_xor(sum, off, 32);
    float inv = 1.0f / sum;
    a_sm[lane] = e0 * inv;
    a_sm[lane + 32] = e1 * inv;
  }
  __syncthreads();

  for (int h = threadIdx.x; h < HID_; h += blockDim.x) {
    const float* eb = enc_out + (size_t)b * S_ * HID_ + h;
    float acc = 0.0f;
#pragma unroll 8
    for (int s = 0; s < S_; ++s) acc += a_sm[s] * eb[(size_t)s * HID_];
    ctx_f32[b * HID_ + h] = acc;
    dec_ctx_bf[(size_t)b * DC_ + HID_ + h] = f2bf(acc);  // ctx half for logits GEMM
  }
}

extern "C" void kernel_launch(void* const* d_in, const int* in_sizes, int n_in,
                              void* d_out, int out_size, void* d_ws, size_t ws_size,
                              hipStream_t stream) {
  const float*         enc_out  = (const float*)d_in[0];
  const unsigned char* src_mask = (const unsigned char*)d_in[1];
  const int*           tgt_in   = (const int*)d_in[2];
  const float*         emb_tab  = (const float*)d_in[3];
  const float*         W_ih0    = (const float*)d_in[4];
  const float*         W_hh0    = (const float*)d_in[5];
  const float*         b_ih0    = (const float*)d_in[6];
  const float*         b_hh0    = (const float*)d_in[7];
  const float*         W_ih1    = (const float*)d_in[8];
  const float*         W_hh1    = (const float*)d_in[9];
  const float*         b_ih1    = (const float*)d_in[10];
  const float*         b_hh1    = (const float*)d_in[11];
  const float*         Wq       = (const float*)d_in[12];
  const float*         Wk       = (const float*)d_in[13];
  const float*         vvec     = (const float*)d_in[14];
  const float*         W_out    = (const float*)d_in[15];
  const float*         b_out    = (const float*)d_in[16];
  float* out = (float*)d_out;

  // ---- workspace carve-up (every size already a multiple of 256 bytes) ----
  char*  ws  = (char*)d_ws;
  size_t off = 0;
  auto carve = [&](size_t bytes) -> char* {
    char* p = ws + off;
    off += (bytes + 255) & ~(size_t)255;
    return p;
  };
  // zero-initialized state block (kept contiguous: one zero kernel)
  float*          h0_f       = (float*)         carve(B_ * HID_ * 4);
  float*          h1_f       = (float*)         carve(B_ * HID_ * 4);
  float*          ctx_f      = (float*)         carve(B_ * HID_ * 4);
  float*          q_f        = (float*)         carve(B_ * HID_ * 4);
  unsigned short* h0_bf      = (unsigned short*)carve(B_ * HID_ * 2);
  unsigned short* dec_ctx_bf = (unsigned short*)carve(B_ * DC_  * 2);
  const long long zero_words = (4LL * B_ * HID_ * 4 + B_ * HID_ * 2 + B_ * DC_ * 2) / 4;
  // scratch
  unsigned short* x_bf    = (unsigned short*)carve((size_t)B_ * XD_ * 2);
  float*          gi      = (float*)         carve((size_t)B_ * H3_ * 4);
  float*          gh      = (float*)         carve((size_t)B_ * H3_ * 4);
  float*          kproj   = (float*)         carve((size_t)B_ * S_ * HID_ * 4);
  unsigned short* enc_bf  = (unsigned short*)carve((size_t)B_ * S_ * HID_ * 2);
  unsigned short* wih0_bf = (unsigned short*)carve((size_t)H3_ * XD_  * 2);
  unsigned short* whh0_bf = (unsigned short*)carve((size_t)H3_ * HID_ * 2);
  unsigned short* wih1_bf = (unsigned short*)carve((size_t)H3_ * HID_ * 2);
  unsigned short* whh1_bf = (unsigned short*)carve((size_t)H3_ * HID_ * 2);
  unsigned short* wq_bf   = (unsigned short*)carve((size_t)HID_ * HID_ * 2);
  unsigned short* wk_bf   = (unsigned short*)carve((size_t)HID_ * HID_ * 2);
  unsigned short* wout_bf = (unsigned short*)carve((size_t)VOCAB_ * DC_ * 2);
  (void)ws_size; (void)in_sizes; (void)n_in; (void)out_size;

  auto conv = [&](const float* src, unsigned short* dst, long long n) {
    int blocks = (int)((n + 255) / 256);
    if (blocks > 4096) blocks = 4096;
    f32_to_bf16_kernel<<<blocks, 256, 0, stream>>>(src, dst, n);
  };
  auto gemm = [&](const unsigned short* A, int lda, const unsigned short* W,
                  const float* bias, float* C, long long ldc, int N, int K,
                  int mblocks) {
    dim3 grid((N / 32 + 7) / 8, mblocks);
    gemm_bf16_wmma<<<grid, 256, 0, stream>>>(A, lda, W, bias, C, ldc, N, K);
  };

  // ---- one-time prep (recomputed every call: deterministic) ----
  zero_u32_kernel<<<(int)((zero_words + 255) / 256), 256, 0, stream>>>(
      (unsigned int*)h0_f, zero_words);
  conv(W_out, wout_bf, (long long)VOCAB_ * DC_);
  conv(W_ih0, wih0_bf, (long long)H3_ * XD_);
  conv(W_hh0, whh0_bf, (long long)H3_ * HID_);
  conv(W_ih1, wih1_bf, (long long)H3_ * HID_);
  conv(W_hh1, whh1_bf, (long long)H3_ * HID_);
  conv(Wq,    wq_bf,   (long long)HID_ * HID_);
  conv(Wk,    wk_bf,   (long long)HID_ * HID_);
  conv(enc_out, enc_bf, (long long)B_ * S_ * HID_);
  // k_proj = enc_out @ Wk^T : M = B*S = 2048 rows
  gemm(enc_bf, HID_, wk_bf, nullptr, kproj, HID_, HID_, HID_, (B_ * S_) / 32);

  // ---- sequential decode loop ----
  for (int t = 0; t < T_; ++t) {
    embed_concat_kernel<<<(B_ * XD_ + 255) / 256, 256, 0, stream>>>(
        emb_tab, tgt_in, ctx_f, x_bf, t);
    // GRU layer 0
    gemm(x_bf, XD_, wih0_bf, b_ih0, gi, H3_, H3_, XD_, 1);
    gemm(h0_bf, HID_, whh0_bf, b_hh0, gh, H3_, H3_, HID_, 1);   // reads h0_{t-1}
    gru_combine_kernel<<<(B_ * HID_ + 255) / 256, 256, 0, stream>>>(
        gi, gh, h0_f, h0_f, h0_bf, HID_);
    // GRU layer 1 (dec_ctx_bf[:, :512] still holds h1_{t-1} in bf16)
    gemm(h0_bf, HID_, wih1_bf, b_ih1, gi, H3_, H3_, HID_, 1);
    gemm(dec_ctx_bf, DC_, whh1_bf, b_hh1, gh, H3_, H3_, HID_, 1);
    gru_combine_kernel<<<(B_ * HID_ + 255) / 256, 256, 0, stream>>>(
        gi, gh, h1_f, h1_f, dec_ctx_bf, DC_);                   // writes dec half
    // attention
    gemm(dec_ctx_bf, DC_, wq_bf, nullptr, q_f, HID_, HID_, HID_, 1);
    attention_kernel<<<B_, 256, 0, stream>>>(
        q_f, kproj, enc_out, src_mask, vvec, ctx_f, dec_ctx_bf); // writes ctx half
    // logits: [32,1024] @ W_out^T -> [32, 32000], strided into [B,T,V]
    gemm(dec_ctx_bf, DC_, wout_bf, b_out, out + (size_t)t * VOCAB_,
         (long long)T_ * VOCAB_, VOCAB_, DC_, 1);
  }
}
